// Mamba_48567490183604
// MI455X (gfx1250) — compile-verified, matrix operand order checked by reading, not run on hardware
//
#include <hip/hip_runtime.h>
#include <hip/hip_bf16.h>
#include <math.h>

// ---- problem constants (match reference) ----
#define Cc   64
#define Dd   128
#define Nn   32
#define Kk   4
#define Rr   4
#define Bb   2
#define Hh   48
#define Wd   48
#define LL   (Hh * Wd)          // 2304
#define XDM  80                 // x_dbl rows padded 68 -> 80 (5 WMMA M-tiles)
#define BL   (Bb * LL)          // 4608 token rows

typedef __bf16 bf16;
typedef __attribute__((ext_vector_type(16))) __bf16 v16bf;
typedef __attribute__((ext_vector_type(8)))  float  v8f;

// ---- workspace layout (bytes) ----
static constexpr size_t WS_XN    = 0;                       // bf16 [BL][Cc]
static constexpr size_t WS_XZ    = WS_XN    + (size_t)BL * Cc * 2;          // f32 [BL][2Dd]
static constexpr size_t WS_XS    = WS_XZ    + (size_t)BL * 2 * Dd * 4;      // f32 [Bb][Kk][Dd][LL]
static constexpr size_t WS_XDBL  = WS_XS    + (size_t)Bb * Kk * Dd * LL * 4;// f32 [Bb][Kk][XDM][LL]
static constexpr size_t WS_BCT   = WS_XDBL  + (size_t)Bb * Kk * XDM * LL * 4;// f32 [Bb][Kk][LL][2Nn]
static constexpr size_t WS_DELTA = WS_BCT   + (size_t)Bb * Kk * LL * 2 * Nn * 4; // f32 [Bb][Kk][Dd][LL]
static constexpr size_t WS_OUTY  = WS_DELTA + (size_t)Bb * Kk * Dd * LL * 4;// f32 [Bb][Kk][Dd][LL]
static constexpr size_t WS_YG    = WS_OUTY  + (size_t)Bb * Kk * Dd * LL * 4;// bf16 [BL][Dd]

// =====================================================================
// 1) channel LayerNorm of fufea1 (B,C,H,W) -> xn bf16 [BL][Cc]
// =====================================================================
__global__ void k_ln(const float* __restrict__ x, const float* __restrict__ g,
                     const float* __restrict__ b, bf16* __restrict__ xn) {
    int p = blockIdx.x * blockDim.x + threadIdx.x;
    if (p >= BL) return;
    int bi = p / LL, pos = p % LL;
    const float* base = x + (size_t)bi * Cc * LL + pos;
    float s = 0.f, ss = 0.f;
    for (int c = 0; c < Cc; ++c) { float v = base[(size_t)c * LL]; s += v; ss += v * v; }
    float mu = s * (1.f / Cc);
    float rs = rsqrtf(ss * (1.f / Cc) - mu * mu + 1e-5f);
    bf16* o = xn + (size_t)p * Cc;
    for (int c = 0; c < Cc; ++c) {
        float v = base[(size_t)c * LL];
        o[c] = (bf16)((v - mu) * rs * g[c] + b[c]);
    }
}

// =====================================================================
// 2) in_proj: xz[BL][256] = xn[BL][64] @ W[256][64]^T   (WMMA bf16)
// =====================================================================
__global__ void k_inproj(const bf16* __restrict__ xn, const float* __restrict__ w,
                         float* __restrict__ xz) {
    int wave = (blockIdx.x * blockDim.x + threadIdx.x) >> 5;
    int lane = threadIdx.x & 31;
    int nt = wave % 16, mt = wave / 16;           // mt < 288
    int row = mt * 16 + (lane & 15);
    int col = nt * 16 + (lane & 15);
    int grp = lane >> 4;
    const bf16*  arow = xn + (size_t)row * Cc;
    const float* wcol = w  + (size_t)col * Cc;
    v8f acc = {};
    for (int kb = 0; kb < Cc; kb += 32) {
        v16bf a, bb;
#pragma unroll
        for (int i = 0; i < 8; ++i) {
            int k0 = kb + grp * 8 + i, k1 = kb + 16 + grp * 8 + i;
            a[i]      = arow[k0];           a[i + 8]  = arow[k1];
            bb[i]     = (bf16)wcol[k0];     bb[i + 8] = (bf16)wcol[k1];
        }
        acc = __builtin_amdgcn_wmma_f32_16x16x32_bf16(false, a, false, bb,
                                                      (short)0, acc, false, false);
    }
#pragma unroll
    for (int j = 0; j < 8; ++j) {
        int m = mt * 16 + grp * 8 + j;
        xz[(size_t)m * (2 * Dd) + col] = acc[j];
    }
}

// =====================================================================
// 3) depthwise 3x3 conv + bias + SiLU, scatter into 4 directions of xs
// =====================================================================
__global__ void k_conv(const float* __restrict__ xz, const float* __restrict__ cw,
                       const float* __restrict__ cb, float* __restrict__ xs) {
    int idx = blockIdx.x * blockDim.x + threadIdx.x;
    if (idx >= Bb * Dd * LL) return;
    int pos = idx % LL; int t = idx / LL; int d = t % Dd; int b = t / Dd;
    int h = pos / Wd, w = pos % Wd;
    float acc = cb[d];
#pragma unroll
    for (int kh = 0; kh < 3; ++kh) {
        int hh = h + kh - 1; if ((unsigned)hh >= (unsigned)Hh) continue;
#pragma unroll
        for (int kw = 0; kw < 3; ++kw) {
            int ww = w + kw - 1; if ((unsigned)ww >= (unsigned)Wd) continue;
            acc += xz[((size_t)b * LL + hh * Wd + ww) * (2 * Dd) + d] * cw[d * 9 + kh * 3 + kw];
        }
    }
    float v = acc / (1.f + __expf(-acc));     // SiLU
    int pwh = w * Hh + h;
    size_t base   = ((size_t)b * Kk) * Dd * LL + (size_t)d * LL;
    size_t stride = (size_t)Dd * LL;
    xs[base + 0 * stride + pos]            = v;
    xs[base + 1 * stride + pwh]            = v;
    xs[base + 2 * stride + (LL - 1 - pos)] = v;
    xs[base + 3 * stride + (LL - 1 - pwh)] = v;
}

// =====================================================================
// 4) x_dbl[b,k] (80xLL, rows<68 valid) = x_proj_w[k](68x128) @ xs[b,k](128xLL)
// =====================================================================
__global__ void k_xdbl(const float* __restrict__ xs, const float* __restrict__ wp,
                       float* __restrict__ xdbl) {
    int wave = (blockIdx.x * blockDim.x + threadIdx.x) >> 5;
    int lane = threadIdx.x & 31;
    int nt = wave % (LL / 16); wave /= (LL / 16);
    int mt = wave % 5;         wave /= 5;
    int k  = wave % Kk;
    int b  = wave / Kk;
    int grp  = lane >> 4;
    int mrow = mt * 16 + (lane & 15);
    int ncol = nt * 16 + (lane & 15);
    bool mvalid = mrow < (Rr + 2 * Nn);
    const float* wrow   = wp + ((size_t)k * (Rr + 2 * Nn) + mrow) * Dd;
    const float* xsbase = xs + (((size_t)b * Kk + k) * Dd) * LL;
    v8f acc = {};
    for (int kb = 0; kb < Dd; kb += 32) {
        v16bf a, bb;
#pragma unroll
        for (int i = 0; i < 8; ++i) {
            int k0 = kb + grp * 8 + i, k1 = kb + 16 + grp * 8 + i;
            a[i]      = (bf16)(mvalid ? wrow[k0] : 0.f);
            a[i + 8]  = (bf16)(mvalid ? wrow[k1] : 0.f);
            bb[i]     = (bf16)xsbase[(size_t)k0 * LL + ncol];
            bb[i + 8] = (bf16)xsbase[(size_t)k1 * LL + ncol];
        }
        acc = __builtin_amdgcn_wmma_f32_16x16x32_bf16(false, a, false, bb,
                                                      (short)0, acc, false, false);
    }
    size_t obase = (((size_t)b * Kk + k) * XDM) * LL;
#pragma unroll
    for (int j = 0; j < 8; ++j) {
        int m = mt * 16 + grp * 8 + j;
        xdbl[obase + (size_t)m * LL + ncol] = acc[j];
    }
}

// =====================================================================
// 5) delta = softplus(dts @ dt_w^T + dt_b)   (R=4, plain VALU)
// =====================================================================
__global__ void k_delta(const float* __restrict__ xdbl, const float* __restrict__ dtw,
                        const float* __restrict__ dtb, float* __restrict__ delta) {
    size_t idx = (size_t)blockIdx.x * blockDim.x + threadIdx.x;
    if (idx >= (size_t)Bb * Kk * Dd * LL) return;
    int l = idx % LL; size_t t = idx / LL;
    int d = t % Dd; t /= Dd; int k = t % Kk; int b = (int)(t / Kk);
    const float* r    = xdbl + (((size_t)b * Kk + k) * XDM) * LL + l;
    const float* wrow = dtw + ((size_t)k * Dd + d) * Rr;
    float acc = dtb[(size_t)k * Dd + d];
#pragma unroll
    for (int rr = 0; rr < Rr; ++rr) acc += r[(size_t)rr * LL] * wrow[rr];
    delta[idx] = (acc > 20.f) ? acc : log1pf(__expf(acc));
}

// =====================================================================
// 6) transpose B,C coeffs: bct[b,k,l,0:64] = xdbl[b,k,4+n,l]
// =====================================================================
__global__ void k_bct(const float* __restrict__ xdbl, float* __restrict__ bct) {
    size_t idx = (size_t)blockIdx.x * blockDim.x + threadIdx.x;
    if (idx >= (size_t)Bb * Kk * 2 * Nn * LL) return;
    int l = idx % LL; size_t t = idx / LL;
    int n = t % (2 * Nn); t /= (2 * Nn); int k = t % Kk; int b = (int)(t / Kk);
    bct[(((size_t)b * Kk + k) * LL + l) * (2 * Nn) + n] =
        xdbl[(((size_t)b * Kk + k) * XDM + Rr + n) * LL + l];
}

// =====================================================================
// 7) selective scan: one wave32 per (b,k,d); lane = state n.
//    32-step blocks: coalesced u/dt loads + shfl broadcast, lane-buffered
//    coalesced output store, forward prefetch of the B/C stream.
//    True serial chain = one FMA per step (exp/reduction are off-chain).
// =====================================================================
__global__ void k_scan(const float* __restrict__ xs, const float* __restrict__ delta,
                       const float* __restrict__ bct, const float* __restrict__ Alog,
                       const float* __restrict__ Dsk, float* __restrict__ outy) {
    int wave = (blockIdx.x * blockDim.x + threadIdx.x) >> 5;
    int lane = threadIdx.x & 31;
    if (wave >= Bb * Kk * Dd) return;
    int d = wave % Dd; int t = wave / Dd; int k = t % Kk; int b = t / Kk;
    float a_n = -__expf(Alog[((size_t)k * Dd + d) * Nn + lane]);
    float dco = Dsk[(size_t)k * Dd + d];
    const float* u  = xs    + (((size_t)b * Kk + k) * Dd + d) * (size_t)LL;
    const float* dt = delta + (((size_t)b * Kk + k) * Dd + d) * (size_t)LL;
    const float* bc = bct   + (((size_t)b * Kk + k) * (size_t)LL) * (2 * Nn);
    float* yo = outy + (((size_t)b * Kk + k) * Dd + d) * (size_t)LL;
    float h = 0.f;
    for (int t0 = 0; t0 < LL; t0 += 32) {
        // one coalesced load of 32 steps of u and dt (lane = step)
        float uv32 = u[t0 + lane];
        float dv32 = dt[t0 + lane];
        const float* bcblk = bc + (size_t)t0 * (2 * Nn);
        // pull the next 32-step B/C tile (8 KB) toward the WGP
        if (t0 + 32 < LL)
            __builtin_prefetch(bc + (size_t)(t0 + 32) * (2 * Nn) + lane * 16, 0, 1);
        float ybuf = 0.f;
#pragma unroll
        for (int j = 0; j < 32; ++j) {
            float uvj = __shfl(uv32, j, 32);
            float dvj = __shfl(dv32, j, 32);
            float Bn = bcblk[(size_t)j * (2 * Nn) + lane];
            float Cn = bcblk[(size_t)j * (2 * Nn) + Nn + lane];
            float dA = __expf(dvj * a_n);
            h = fmaf(dvj * uvj, Bn, dA * h);      // <- only serial dependency
            float p = h * Cn;
#pragma unroll
            for (int off = 16; off > 0; off >>= 1) p += __shfl_xor(p, off, 32);
            p = fmaf(dco, uvj, p);
            if (lane == j) ybuf = p;              // lane j keeps step j's output
        }
        yo[t0 + lane] = ybuf;                     // one coalesced 128B store / 32 steps
    }
}

// =====================================================================
// 8) merge 4 directions, out-LN over D, gate with SiLU(z) -> yg bf16
// =====================================================================
__global__ void k_merge(const float* __restrict__ outy, const float* __restrict__ xz,
                        const float* __restrict__ og, const float* __restrict__ ob,
                        bf16* __restrict__ yg) {
    int wave = (blockIdx.x * blockDim.x + threadIdx.x) >> 5;
    int lane = threadIdx.x & 31;
    if (wave >= BL) return;
    int b = wave / LL, pos = wave % LL;
    int h = pos / Wd, w = pos % Wd;
    int pwh = w * Hh + h;
    size_t base = ((size_t)b * Kk) * Dd * LL;
    size_t st   = (size_t)Dd * LL;
    float v[4]; float s = 0.f;
#pragma unroll
    for (int i = 0; i < 4; ++i) {
        int d = lane + i * 32;
        size_t dr = (size_t)d * LL;
        float y = outy[base + 0 * st + dr + pos]
                + outy[base + 1 * st + dr + pwh]
                + outy[base + 2 * st + dr + (LL - 1 - pos)]
                + outy[base + 3 * st + dr + (LL - 1 - pwh)];
        v[i] = y; s += y;
    }
#pragma unroll
    for (int off = 16; off > 0; off >>= 1) s += __shfl_xor(s, off, 32);
    float mu = s * (1.f / Dd);
    float ss = 0.f;
#pragma unroll
    for (int i = 0; i < 4; ++i) { float dlt = v[i] - mu; ss += dlt * dlt; }
#pragma unroll
    for (int off = 16; off > 0; off >>= 1) ss += __shfl_xor(ss, off, 32);
    float rs = rsqrtf(ss * (1.f / Dd) + 1e-5f);
    const float* zrow = xz + ((size_t)b * LL + pos) * (2 * Dd) + Dd;
    bf16* orow = yg + ((size_t)b * LL + pos) * Dd;
#pragma unroll
    for (int i = 0; i < 4; ++i) {
        int d = lane + i * 32;
        float z  = zrow[d];
        float gz = z / (1.f + __expf(-z));
        float yn = (v[i] - mu) * rs * og[d] + ob[d];
        orow[d] = (bf16)(yn * gz);
    }
}

// =====================================================================
// 9) out_proj + residual: out[B,C,H,W] = yg(BLx128) @ Wout(64x128)^T + fufea1
// =====================================================================
__global__ void k_outproj(const bf16* __restrict__ yg, const float* __restrict__ wout,
                          const float* __restrict__ resid, float* __restrict__ out) {
    int wave = (blockIdx.x * blockDim.x + threadIdx.x) >> 5;
    int lane = threadIdx.x & 31;
    int nt = wave % 4, mt = wave / 4;            // mt < 288
    int grp = lane >> 4;
    int row = mt * 16 + (lane & 15);
    int col = nt * 16 + (lane & 15);
    const bf16*  arow = yg   + (size_t)row * Dd;
    const float* wcol = wout + (size_t)col * Dd;
    v8f acc = {};
    for (int kb = 0; kb < Dd; kb += 32) {
        v16bf a, bb;
#pragma unroll
        for (int i = 0; i < 8; ++i) {
            int k0 = kb + grp * 8 + i, k1 = kb + 16 + grp * 8 + i;
            a[i]      = arow[k0];        a[i + 8]  = arow[k1];
            bb[i]     = (bf16)wcol[k0];  bb[i + 8] = (bf16)wcol[k1];
        }
        acc = __builtin_amdgcn_wmma_f32_16x16x32_bf16(false, a, false, bb,
                                                      (short)0, acc, false, false);
    }
#pragma unroll
    for (int j = 0; j < 8; ++j) {
        int m = mt * 16 + grp * 8 + j;
        int b = m / LL, pos = m % LL;
        size_t o = ((size_t)b * Cc + col) * (size_t)LL + pos;
        out[o] = acc[j] + resid[o];
    }
}

// =====================================================================
extern "C" void kernel_launch(void* const* d_in, const int* in_sizes, int n_in,
                              void* d_out, int out_size, void* d_ws, size_t ws_size,
                              hipStream_t stream) {
    (void)in_sizes; (void)n_in; (void)out_size; (void)ws_size;
    const float* fufea1   = (const float*)d_in[0];
    const float* ln_g     = (const float*)d_in[1];
    const float* ln_b     = (const float*)d_in[2];
    const float* in_projw = (const float*)d_in[3];
    const float* conv_w   = (const float*)d_in[4];
    const float* conv_b   = (const float*)d_in[5];
    const float* x_projw  = (const float*)d_in[6];
    const float* dt_w     = (const float*)d_in[7];
    const float* dt_b     = (const float*)d_in[8];
    const float* A_logs   = (const float*)d_in[9];
    const float* Ds       = (const float*)d_in[10];
    const float* out_ln_g = (const float*)d_in[11];
    const float* out_ln_b = (const float*)d_in[12];
    const float* out_projw= (const float*)d_in[13];
    float* out = (float*)d_out;

    char* ws = (char*)d_ws;
    bf16*  xn    = (bf16*) (ws + WS_XN);
    float* xz    = (float*)(ws + WS_XZ);
    float* xs    = (float*)(ws + WS_XS);
    float* xdbl  = (float*)(ws + WS_XDBL);
    float* bct   = (float*)(ws + WS_BCT);
    float* delta = (float*)(ws + WS_DELTA);
    float* outy  = (float*)(ws + WS_OUTY);
    bf16*  yg    = (bf16*) (ws + WS_YG);

    // 1) LN
    k_ln<<<(BL + 255) / 256, 256, 0, stream>>>(fufea1, ln_g, ln_b, xn);
    // 2) in_proj (4608 waves, 4 waves/block)
    k_inproj<<<(BL / 16) * 16 / 4, 128, 0, stream>>>(xn, in_projw, xz);
    // 3) conv + silu + direction scatter
    k_conv<<<(Bb * Dd * LL + 255) / 256, 256, 0, stream>>>(xz, conv_w, conv_b, xs);
    // 4) x_dbl GEMMs (Bb*Kk*5*(LL/16) = 5760 waves)
    k_xdbl<<<(Bb * Kk * 5 * (LL / 16)) / 4, 128, 0, stream>>>(xs, x_projw, xdbl);
    // 5) delta
    k_delta<<<(int)(((size_t)Bb * Kk * Dd * LL + 255) / 256), 256, 0, stream>>>(xdbl, dt_w, dt_b, delta);
    // 6) B/C transpose
    k_bct<<<(int)(((size_t)Bb * Kk * 2 * Nn * LL + 255) / 256), 256, 0, stream>>>(xdbl, bct);
    // 7) selective scan (1024 waves, 8/block)
    k_scan<<<(Bb * Kk * Dd) / 8, 256, 0, stream>>>(xs, delta, bct, A_logs, Ds, outy);
    // 8) merge + out-LN + gate (4608 waves, 8/block)
    k_merge<<<BL / 8, 256, 0, stream>>>(outy, xz, out_ln_g, out_ln_b, yg);
    // 9) out_proj + residual (1152 waves, 4/block)
    k_outproj<<<(288 * 4) / 4, 128, 0, stream>>>(yg, out_projw, fufea1, out);
}
